// GCN_37812892074319
// MI455X (gfx1250) — matrix operand order repaired and checked by reading, not compiled, not run on hardware
//
#include <hip/hip_runtime.h>

// ---------------------------------------------------------------------------
// Types for CDNA5 WMMA (wave32, 16x16x32 bf16 -> f32 accumulate)
// ---------------------------------------------------------------------------
typedef __bf16 bf16_t;
typedef __attribute__((ext_vector_type(16))) __bf16 v16bf;
typedef __attribute__((ext_vector_type(8)))  float  v8f;

#define CH 128
#define NODES_PER_TILE 128

// Async copy of one 16-byte chunk: global -> LDS (ASYNCcnt-tracked).
// %0 = VGPR with LDS byte address (wave LDS base-relative),
// %1 = VGPR pair with 64-bit global address.
#define ASYNC_CP16(ldsoff, gaddr)                                   \
    asm volatile("global_load_async_to_lds_b128 %0, %1, off"        \
                 :: "v"(ldsoff), "v"(gaddr) : "memory")

// ---------------------------------------------------------------------------
// Pre-split weights (run once per call): W[128][128] fp32 (row-major [k][n])
//   -> WhiT[n][k], WloT[n][k] bf16, stored contiguously (32KB hi + 32KB lo)
// Handles `nmat` matrices stacked contiguously.
// ---------------------------------------------------------------------------
__global__ __launch_bounds__(256) void split_weights_kernel(
    const float* __restrict__ W, bf16_t* __restrict__ out, int nmat)
{
    const int i = blockIdx.x * blockDim.x + threadIdx.x;
    if (i >= nmat * CH * CH) return;
    const int m   = i >> 14;
    const int rem = i & 16383;
    const int k   = rem >> 7;
    const int n   = rem & 127;
    const float w  = W[i];                       // W[m][k][n]
    const bf16_t h = (bf16_t)w;
    bf16_t* o = out + (size_t)m * 2 * CH * CH + (size_t)n * CH + k;
    o[0]         = h;                            // hi plane
    o[CH * CH]   = (bf16_t)(w - (float)h);       // lo plane
}

// ---------------------------------------------------------------------------
// GEMM: C[n x 128] = op( A[n x 128] @ W[128 x 128] + bias )
// MODE 0: C  = A@W + b            MODE 1: C = lrelu(A@W + b)
// MODE 2: C += A@W + b
//
// Wsplit: pre-split transposed bf16 weights (hi 32KB then lo 32KB).
// One block = 8 waves = one 128x128 C tile; K=128.
//   - W staged into LDS via global_load_async_to_lds_b128 (64KB, offset 0)
//   - A fragments loaded straight from global fp32, hi/lo split in registers
// ---------------------------------------------------------------------------
template <int MODE>
__global__ __launch_bounds__(256) void gemm128_wmma(
    const float* __restrict__ A, const bf16_t* __restrict__ Wsplit,
    const float* __restrict__ bias, float* __restrict__ C, int n_rows)
{
    // Only LDS object in this kernel -> allocated at LDS offset 0.
    __shared__ bf16_t sB[2][CH][CH];   // [hi/lo][n][k], 64KB

    const int tid  = threadIdx.x;      // 0..255
    const int wave = tid >> 5;         // 0..7
    const int lane = tid & 31;
    const int row0 = blockIdx.x * NODES_PER_TILE;

    // ---- async-stage pre-split W: 64KB, 256B per thread (16 x b128) ----
    {
        const unsigned base = (unsigned)tid * 256u;
        const unsigned long long gbase =
            (unsigned long long)(uintptr_t)Wsplit + (unsigned long long)base;
        #pragma unroll
        for (int c = 0; c < 16; ++c) {
            const unsigned          lo = base + (unsigned)(c * 16);
            const unsigned long long ga = gbase + (unsigned long long)(c * 16);
            ASYNC_CP16(lo, ga);
        }
    }

    // Prefetch next block's A tile into L2 (64KB = 512 cachelines)
    {
        const int nrow0 = row0 + NODES_PER_TILE;
        if (nrow0 < n_rows) {
            const float* nxt = A + (size_t)nrow0 * CH;
            for (int i = tid; i < 512; i += 256)
                __builtin_prefetch(nxt + i * 32, 0, 1);
        }
    }

    asm volatile("s_wait_asynccnt 0" ::: "memory");
    __syncthreads();

    // ---- per-wave fragment addressing (ISA 7.12.2 layouts) ----
    const int m0    = wave * 16;
    const int aRowG = row0 + m0 + (lane & 15);
    const int aKoff = (lane >> 4) << 3;      // lanes 16-31: K runs start at +8
    const int bN    = lane & 15;
    const int bKoff = (lane >> 4) << 3;      // lanes 16-31: K runs start at +8

    // Branchless row guard (keeps EXEC all-1s for WMMA)
    const bool  valid = aRowG < n_rows;
    const float scale = valid ? 1.0f : 0.0f;
    const float* arow = A + (size_t)(valid ? aRowG : 0) * CH;

    v8f acc[8] = {};

    #pragma unroll
    for (int kk = 0; kk < CH; kk += 32) {
        // A fragment straight from global: two 32B runs (k+0..7, k+16..23)
        float af[16];
        {
            const float4 f0 = *(const float4*)(arow + kk + aKoff);
            const float4 f1 = *(const float4*)(arow + kk + aKoff + 4);
            const float4 f2 = *(const float4*)(arow + kk + aKoff + 16);
            const float4 f3 = *(const float4*)(arow + kk + aKoff + 20);
            af[0]=f0.x; af[1]=f0.y; af[2]=f0.z; af[3]=f0.w;
            af[4]=f1.x; af[5]=f1.y; af[6]=f1.z; af[7]=f1.w;
            af[8]=f2.x; af[9]=f2.y; af[10]=f2.z; af[11]=f2.w;
            af[12]=f3.x; af[13]=f3.y; af[14]=f3.z; af[15]=f3.w;
        }
        v16bf ahi, alo;
        #pragma unroll
        for (int i = 0; i < 16; ++i) {
            const float a  = af[i] * scale;
            const bf16_t h = (bf16_t)a;
            ahi[i] = h;
            alo[i] = (bf16_t)(a - (float)h);
        }

        #pragma unroll
        for (int t = 0; t < 8; ++t) {
            const int n0 = t * 16;
            v16bf bhi, blo;
            #pragma unroll
            for (int i = 0; i < 16; ++i) {
                bhi[i] = sB[0][n0 + bN][kk + bKoff + (i & 7) + ((i >> 3) << 4)];  // two 16B runs (k+0..7, k+16..23)
                blo[i] = sB[1][n0 + bN][kk + bKoff + (i & 7) + ((i >> 3) << 4)];
            }
            acc[t] = __builtin_amdgcn_wmma_f32_16x16x32_bf16(
                false, ahi, false, bhi, (short)0, acc[t], false, false);
            acc[t] = __builtin_amdgcn_wmma_f32_16x16x32_bf16(
                false, ahi, false, blo, (short)0, acc[t], false, false);
            acc[t] = __builtin_amdgcn_wmma_f32_16x16x32_bf16(
                false, alo, false, bhi, (short)0, acc[t], false, false);
        }
    }

    // ---- epilogue: D layout: N = lane&15, lanes>=16 add 8 to M, VGPR v = M+v
    const int cn    = lane & 15;
    const int mhalf = (lane >> 4) << 3;
    #pragma unroll
    for (int t = 0; t < 8; ++t) {
        const int n0 = t * 16;
        const float b = bias[n0 + cn];
        #pragma unroll
        for (int v = 0; v < 8; ++v) {
            const int gr = row0 + m0 + mhalf + v;
            if (gr < n_rows) {
                float val = acc[t][v] + b;
                float* p = &C[(size_t)gr * CH + n0 + cn];
                if (MODE == 1) val = (val > 0.0f) ? val : 0.01f * val;
                if (MODE == 2) val += *p;
                *p = val;
            }
        }
    }
}

// ---------------------------------------------------------------------------
// Edge scatter: out[dst] += y[src]  (one wave per edge, float4 per lane)
// ---------------------------------------------------------------------------
__global__ __launch_bounds__(256) void scatter_add_kernel(
    const float* __restrict__ y, const int* __restrict__ src,
    const int* __restrict__ dst, float* __restrict__ out, int n_edges)
{
    const int gwave   = (int)((blockIdx.x * blockDim.x + threadIdx.x) >> 5);
    const int lane    = threadIdx.x & 31;
    const int n_waves = (int)((gridDim.x * blockDim.x) >> 5);
    for (int e = gwave; e < n_edges; e += n_waves) {
        const int s = src[e];
        const int d = dst[e];
        const float4 v = ((const float4*)(y + (size_t)s * CH))[lane];
        float* p = out + (size_t)d * CH + lane * 4;
        atomicAdd(p + 0, v.x);
        atomicAdd(p + 1, v.y);
        atomicAdd(p + 2, v.z);
        atomicAdd(p + 3, v.w);
    }
}

// ---------------------------------------------------------------------------
// Elementwise kernels (float4 vectorized)
// ---------------------------------------------------------------------------
__global__ __launch_bounds__(256) void lrelu_kernel(
    const float* __restrict__ in, float* __restrict__ out, int n4)
{
    int i = blockIdx.x * blockDim.x + threadIdx.x;
    if (i < n4) {
        float4 v = ((const float4*)in)[i];
        v.x = v.x > 0.0f ? v.x : 0.01f * v.x;
        v.y = v.y > 0.0f ? v.y : 0.01f * v.y;
        v.z = v.z > 0.0f ? v.z : 0.01f * v.z;
        v.w = v.w > 0.0f ? v.w : 0.01f * v.w;
        ((float4*)out)[i] = v;
    }
}

__global__ __launch_bounds__(256) void add_inplace_kernel(
    float* __restrict__ h, const float* __restrict__ a, int n4)
{
    int i = blockIdx.x * blockDim.x + threadIdx.x;
    if (i < n4) {
        float4 x = ((float4*)h)[i];
        const float4 y = ((const float4*)a)[i];
        x.x += y.x; x.y += y.y; x.z += y.z; x.w += y.w;
        ((float4*)h)[i] = x;
    }
}

// ---------------------------------------------------------------------------
// Orchestration
// ---------------------------------------------------------------------------
extern "C" void kernel_launch(void* const* d_in, const int* in_sizes, int n_in,
                              void* d_out, int out_size, void* d_ws, size_t ws_size,
                              hipStream_t stream)
{
    const float* x      = (const float*)d_in[0];
    const int*   ei     = (const int*)  d_in[1];
    const float* w_in   = (const float*)d_in[2];
    const float* b_in   = (const float*)d_in[3];
    const float* taps_w = (const float*)d_in[4];   // [2][4][128][128]
    const float* taps_b = (const float*)d_in[5];   // [2][4][128]
    const float* w_out  = (const float*)d_in[6];
    const float* b_out  = (const float*)d_in[7];
    float*       outp   = (float*)d_out;

    const int N = in_sizes[0] / CH;        // 50000 nodes
    const int E = in_sizes[1] / 2;         // 800000 edges
    const int* src = ei;
    const int* dst = ei + E;

    // Workspace layout:
    //   [0]           10 pre-split weight images, 64KB each (hi+lo, transposed)
    //   [wsplit_end]  h, y, yn, acc feature buffers (fp32, N*128 each)
    const size_t WSPLIT = (size_t)2 * CH * CH;         // bf16 elems per matrix
    bf16_t* wsp   = (bf16_t*)d_ws;
    bf16_t* ws_in  = wsp;                               // w_in
    bf16_t* ws_tap = wsp + WSPLIT;                      // taps[0..7]
    bf16_t* ws_out = wsp + 9 * WSPLIT;                  // w_out
    float*  h   = (float*)(wsp + 10 * WSPLIT);
    const size_t NF = (size_t)N * CH;
    float* y   = h  + NF;
    float* yn  = y  + NF;
    float* acc = yn + NF;

    const int gemm_blocks = (N + NODES_PER_TILE - 1) / NODES_PER_TILE;  // 391
    const int ew_blocks   = (int)((NF / 4 + 255) / 256);
    const int sc_blocks   = (E + 7) / 8;   // one wave per edge

    // Pre-split all weights once (w_in: 1 mat, taps: 8 mats, w_out: 1 mat)
    split_weights_kernel<<<(1 * CH * CH + 255) / 256, 256, 0, stream>>>(w_in,  ws_in,  1);
    split_weights_kernel<<<(8 * CH * CH + 255) / 256, 256, 0, stream>>>(taps_w, ws_tap, 8);
    split_weights_kernel<<<(1 * CH * CH + 255) / 256, 256, 0, stream>>>(w_out, ws_out, 1);

    // readin: h = lrelu(x @ w_in + b_in)
    gemm128_wmma<1><<<gemm_blocks, 256, 0, stream>>>(x, ws_in, b_in, h, N);

    for (int l = 0; l < 2; ++l) {
        bf16_t*      Wl = ws_tap + (size_t)l * 4 * WSPLIT;
        const float* Bl = taps_b + (size_t)l * 4 * CH;

        // y = act(h); acc = y @ W0 + b0
        lrelu_kernel<<<ew_blocks, 256, 0, stream>>>(h, y, (int)(NF / 4));
        gemm128_wmma<0><<<gemm_blocks, 256, 0, stream>>>(y, Wl, Bl, acc, N);

        float* cur = y;
        float* nxt = yn;
        for (int k = 1; k <= 3; ++k) {
            hipMemsetAsync(nxt, 0, NF * sizeof(float), stream);
            scatter_add_kernel<<<sc_blocks, 256, 0, stream>>>(cur, src, dst, nxt, E);
            gemm128_wmma<2><<<gemm_blocks, 256, 0, stream>>>(
                nxt, Wl + (size_t)k * WSPLIT, Bl + (size_t)k * CH, acc, N);
            float* t = cur; cur = nxt; nxt = t;
        }
        // h += acc
        add_inplace_kernel<<<ew_blocks, 256, 0, stream>>>(h, acc, (int)(NF / 4));
    }

    // readout: out = h @ w_out + b_out
    gemm128_wmma<0><<<gemm_blocks, 256, 0, stream>>>(h, ws_out, b_out, outp, N);
}
